// SSIMMetric_1125281431818
// MI455X (gfx1250) — matrix operand order repaired and checked by reading, not compiled, not run on hardware
//
#include <hip/hip_runtime.h>
#include <hip/hip_bf16.h>

typedef __attribute__((ext_vector_type(16))) _Float16 v16h;
typedef __attribute__((ext_vector_type(8)))  float    v8f;

#define WIN   11
#define RAD   5
#define IMH   512
#define IMW   512
#define NBC   96          /* 32 batch * 3 channels */
#define REGION 64         /* 64x64 output region per workgroup */
#define SROWS 80          /* staged rows  = REGION + 2*(RAD) + extra A2 slack */
#define SCOLS 80          /* staged cols  = REGION + RAD + (16-RAD) */
#define SLDP  81          /* padded LDS row stride (odd vs 64 banks) */
#define SSIM_C1 0.0001f   /* 0.01^2 */
#define SSIM_C2 0.0009f   /* 0.03^2 */

__device__ __forceinline__ float gauss_unnorm(int i) {
    float x = (float)(i - RAD);
    return __expf(-x * x / (2.0f * 1.5f * 1.5f));
}

__global__ __launch_bounds__(256) void ssim_main_kernel(const float* __restrict__ img1,
                                                        const float* __restrict__ img2,
                                                        float* __restrict__ partial)
{
    __shared__ float s1[SROWS * SLDP];
    __shared__ float s2[SROWS * SLDP];
    __shared__ float wsum[8];

    const int tid  = threadIdx.x;
    const int lane = tid & 31;
    const int wave = tid >> 5;

    const int bc  = blockIdx.x >> 6;        /* which channel-image (0..95)   */
    const int t64 = blockIdx.x & 63;        /* which 64x64 region (8x8 grid) */
    const int r0  = (t64 >> 3) * REGION;
    const int c0  = (t64 & 7)  * REGION;

    const float* im1 = img1 + (size_t)bc * IMH * IMW;
    const float* im2 = img2 + (size_t)bc * IMH * IMW;

    /* normalized Gaussian taps (computed in fp32, unrolled, no scratch array) */
    float gsum = 0.f;
#pragma unroll
    for (int i = 0; i < WIN; ++i) gsum += gauss_unnorm(i);
    const float ginv = 1.0f / gsum;

    /* ---- stage 80x80 halo blocks of both images into LDS (zero-padded) ---- */
    for (int idx = tid; idx < SROWS * SCOLS; idx += 256) {
        const int r  = idx / SCOLS;
        const int c  = idx - r * SCOLS;
        const int gr = r0 - RAD + r;
        const int gc = c0 - RAD + c;
        const bool inb = (gr >= 0) & (gr < IMH) & (gc >= 0) & (gc < IMW);
        const size_t off = (size_t)gr * IMW + gc;
        s1[r * SLDP + c] = inb ? im1[off] : 0.0f;
        s2[r * SLDP + c] = inb ? im2[off] : 0.0f;
    }
    __syncthreads();

    const int m    = lane & 15;   /* row (A) / col (B,C/D) within tile */
    const int half = lane >> 4;   /* lane-half selects K/M sub-blocks  */

    /* ---- constant banded-Gaussian fragments ----
       Gfrag (A, 16x32): lane element e -> K = e + 16*half
       KbB (B, 32x16): lane element e -> K = e + 8*half + 8*(e/8), N = lane%16 */
    v16h Gfrag, KbB;
#pragma unroll
    for (int e = 0; e < 16; ++e) {
        const int kA = e + 16 * half;
        const int dA = kA - m;
        Gfrag[e] = (_Float16)((dA >= 0 && dA < WIN) ? gauss_unnorm(dA) * ginv : 0.0f);
        const int kB = e + 8 * half + 8 * (e >> 3);
        const int dB = kB - m;
        KbB[e]   = (_Float16)((dB >= 0 && dB < WIN) ? gauss_unnorm(dB) * ginv : 0.0f);
    }

    float lsum = 0.0f;

    /* 16 tiles of 16x16 per region; wave w handles tiles w and w+8 */
    for (int t = wave; t < 16; t += 8) {
        const int ty = t >> 2, tx = t & 3;
        const int rowA1   = ty * 16 + m;        /* staged row of A1 (= tile_r-5+m) */
        const int rowA2   = rowA1 + 16;
        const int colbase = tx * 16;

        /* raw fp32 tile rows for both images (A1 = K-rows 0..15, A2 = 16..31) */
        float r1a[16], r2a[16], r1b[16], r2b[16];
#pragma unroll
        for (int e = 0; e < 16; ++e) {
            const int k = e + 8 * half + 8 * (e >> 3);
            const int c = colbase + k;
            r1a[e] = s1[rowA1 * SLDP + c];
            r2a[e] = s2[rowA1 * SLDP + c];
            r1b[e] = s1[rowA2 * SLDP + c];
            r2b[e] = s2[rowA2 * SLDP + c];
        }

        const v8f zc = {};
        v8f acc[5];

#pragma unroll
        for (int q = 0; q < 5; ++q) {
            /* build f16 A fragments for quantity q */
            v16h a1, a2;
#pragma unroll
            for (int e = 0; e < 16; ++e) {
                float va, vb;
                if      (q == 0) { va = r1a[e];          vb = r1b[e]; }
                else if (q == 1) { va = r2a[e];          vb = r2b[e]; }
                else if (q == 2) { va = r1a[e] * r1a[e]; vb = r1b[e] * r1b[e]; }
                else if (q == 3) { va = r2a[e] * r2a[e]; vb = r2b[e] * r2b[e]; }
                else             { va = r1a[e] * r2a[e]; vb = r1b[e] * r2b[e]; }
                a1[e] = (_Float16)va;
                a2[e] = (_Float16)vb;
            }

            /* horizontal pass: rows -5..+10 and +11..+26 of the tile */
            v8f h1 = __builtin_amdgcn_wmma_f32_16x16x32_f16(false, a1, false, KbB,
                                                            (short)0, zc, false, false);
            v8f h2 = __builtin_amdgcn_wmma_f32_16x16x32_f16(false, a2, false, KbB,
                                                            (short)0, zc, false, false);

            /* C(f32)-layout -> B(f16)-layout: needs only a half-swap shuffle */
            v16h bv;
#pragma unroll
            for (int v = 0; v < 8; ++v) {
                const int p0 = (2 * v) & 7;
                const int p1 = (2 * v + 1) & 7;
                float k0, k1;
                if (v < 4) {                 /* source rows live in lanes 0..15 */
                    const float o0 = __shfl_xor(h2[p0], 16, 32);
                    const float o1 = __shfl_xor(h2[p1], 16, 32);
                    k0 = half ? o0 : h1[p0];
                    k1 = half ? o1 : h1[p1];
                } else {                     /* source rows live in lanes 16..31 */
                    const float o0 = __shfl_xor(h1[p0], 16, 32);
                    const float o1 = __shfl_xor(h1[p1], 16, 32);
                    k0 = half ? h2[p0] : o0;
                    k1 = half ? h2[p1] : o1;
                }
                bv[2 * v]     = (_Float16)k0;
                bv[2 * v + 1] = (_Float16)k1;
            }

            /* vertical pass: banded G as A operand */
            acc[q] = __builtin_amdgcn_wmma_f32_16x16x32_f16(false, Gfrag, false, bv,
                                                            (short)0, zc, false, false);
        }

        /* pointwise SSIM on the 8 accumulator elements per lane */
#pragma unroll
        for (int p = 0; p < 8; ++p) {
            const float mu1  = acc[0][p];
            const float mu2  = acc[1][p];
            const float mu1s = mu1 * mu1;
            const float mu2s = mu2 * mu2;
            const float mu12 = mu1 * mu2;
            const float sg1  = acc[2][p] - mu1s;
            const float sg2  = acc[3][p] - mu2s;
            const float sg12 = acc[4][p] - mu12;
            const float num  = (2.0f * mu12 + SSIM_C1) * (2.0f * sg12 + SSIM_C2);
            const float den  = (mu1s + mu2s + SSIM_C1) * (sg1 + sg2 + SSIM_C2);
            lsum += num / den;
        }
    }

    /* deterministic wave -> block reduction */
#pragma unroll
    for (int o = 16; o > 0; o >>= 1) lsum += __shfl_xor(lsum, o, 32);
    if (lane == 0) wsum[wave] = lsum;
    __syncthreads();
    if (tid == 0) {
        float s = 0.0f;
#pragma unroll
        for (int w = 0; w < 8; ++w) s += wsum[w];
        partial[blockIdx.x] = s;
    }
}

__global__ __launch_bounds__(256) void ssim_reduce_kernel(const float* __restrict__ partial,
                                                          float* __restrict__ out, int n)
{
    __shared__ float sm[256];
    float s = 0.0f;
    for (int i = threadIdx.x; i < n; i += 256) s += partial[i];
    sm[threadIdx.x] = s;
    __syncthreads();
    for (int off = 128; off > 0; off >>= 1) {
        if ((int)threadIdx.x < off) sm[threadIdx.x] += sm[threadIdx.x + off];
        __syncthreads();
    }
    if (threadIdx.x == 0)
        out[0] = sm[0] * (1.0f / ((float)NBC * (float)IMH * (float)IMW));
}

extern "C" void kernel_launch(void* const* d_in, const int* in_sizes, int n_in,
                              void* d_out, int out_size, void* d_ws, size_t ws_size,
                              hipStream_t stream)
{
    (void)in_sizes; (void)n_in; (void)out_size; (void)ws_size;
    const float* img1 = (const float*)d_in[0];
    const float* img2 = (const float*)d_in[1];
    float* out     = (float*)d_out;
    float* partial = (float*)d_ws;            /* 6144 floats = 24 KB scratch */

    const int nblocks = NBC * (IMH / REGION) * (IMW / REGION);   /* 96*64 = 6144 */
    ssim_main_kernel<<<nblocks, 256, 0, stream>>>(img1, img2, partial);
    ssim_reduce_kernel<<<1, 256, 0, stream>>>(partial, out, nblocks);
}
